// ActionNet_48679159333357
// MI455X (gfx1250) — matrix operand order repaired and checked by reading, not compiled, not run on hardware
//
#include <hip/hip_runtime.h>
#include <hip/hip_bf16.h>
#include <math.h>

// ---------------------------------------------------------------------------
// GATv2 x8 stack for MI455X (gfx1250): bf16 WMMA GEMMs (transposed weight
// panels for contiguous B fragments) + fused edge kernels.
// ---------------------------------------------------------------------------

#define GAT_N 1024
#define GAT_E 8192
#define GAT_H 32
#define NEG_SLOPE 0.2f

typedef __attribute__((ext_vector_type(16))) __bf16 v16bf;
typedef __attribute__((ext_vector_type(8)))  __bf16 v8bf;
typedef __attribute__((ext_vector_type(8)))  float  v8f;

static __device__ __forceinline__ v16bf cat16(v8bf lo, v8bf hi) {
  return __builtin_shufflevector(lo, hi, 0, 1, 2, 3, 4, 5, 6, 7,
                                 8, 9, 10, 11, 12, 13, 14, 15);
}

// ---------------- conversion / fill kernels --------------------------------

// X (rows x cols f32) -> (rows x colsPad bf16), zero-pad extra columns.
__global__ void k_cvt_pad_cols(const float* __restrict__ in, __bf16* __restrict__ out,
                               int rows, int cols, int colsPad) {
  int idx = blockIdx.x * blockDim.x + threadIdx.x;
  if (idx >= rows * colsPad) return;
  int r = idx / colsPad, c = idx - r * colsPad;
  out[idx] = (c < cols) ? (__bf16)in[r * cols + c] : (__bf16)0.0f;
}

// W (K x M f32, row-major) -> WT (M x Kpad bf16): WT[j*Kpad + k] = W[k*M + j].
// Tiled 32x32 transpose through LDS; both global accesses coalesced.
// Requires M % 32 == 0 (true for all layers: min M = 32).
__global__ void k_cvt_w_transpose(const float* __restrict__ in, __bf16* __restrict__ out,
                                  int K, int Kpad, int M) {
  __shared__ float tile[32][33];
  const int jBlk = blockIdx.x * 32;
  const int kBlk = blockIdx.y * 32;
  const int tx = threadIdx.x & 31;
  const int ty = threadIdx.x >> 5;          // 8 row-groups
#pragma unroll
  for (int r = ty; r < 32; r += 8) {        // read: coalesced along j
    int k = kBlk + r;
    tile[r][tx] = (k < K) ? in[(size_t)k * M + jBlk + tx] : 0.0f;
  }
  __syncthreads();
#pragma unroll
  for (int r = ty; r < 32; r += 8) {        // write: coalesced along k
    out[(size_t)(jBlk + r) * Kpad + kBlk + tx] = (__bf16)tile[tx][r];
  }
}

__global__ void k_fill(float* __restrict__ p, float v, int n) {
  int idx = blockIdx.x * blockDim.x + threadIdx.x;
  if (idx < n) p[idx] = v;
}

// ---------------- bf16 WMMA GEMM:  Out[NxM] = X[NxKpad] @ W ------------------
// W supplied as transposed panels WT[M x Kpad] so every fragment load is a
// contiguous, aligned wide load with constant-stride (64B) k-advance.
// Block = 128 threads = 4 wave32s; each wave owns TWO 16x16 output tiles
// (shared A fragment, independent accumulator chains).
// grid = (ceil(M/128), N/16). A/B share a contiguous k order; D layout per CDNA5 ISA 7.12.2.

__global__ void k_gemm_bf16(const __bf16* __restrict__ X,    // N x Kpad
                            const __bf16* __restrict__ WT,   // M x Kpad
                            float* __restrict__ Out,         // N x M
                            int Kpad, int M) {
  const int lane = threadIdx.x & 31;
  const int wave = threadIdx.x >> 5;
  const int j0 = blockIdx.x * 128 + wave * 32;
  if (j0 >= M) return;                      // wave-uniform: EXEC stays all-ones
  const int i0 = blockIdx.y * 16;

  const int half = (lane >> 4) & 1;         // lane half-select
  const int arow = i0 + (lane & 15);        // A: lane -> output row
  const int jcol = j0 + (lane & 15);        // B/D: lane -> output col

  // A: lanes 0-15 K = kb..kb+15; lanes 16-31 K = kb+16..kb+31.
  const __bf16* ap  = X + (size_t)arow * Kpad + half * 16;
  // B: lanes 0-15 hold K = kb..kb+15, lanes 16-31 hold K = kb+16..kb+31,
  //    for column jcol -> contiguous 32B in WT.
  const __bf16* bp0 = WT + (size_t)jcol * Kpad + half * 16;
  const __bf16* bp1 = bp0 + (size_t)16 * Kpad;

  v8f acc0 = {}, acc1 = {};
  for (int kb = 0; kb < Kpad; kb += 32) {
    v16bf a  = *(const v16bf*)(ap + kb);
    v16bf b0 = *(const v16bf*)(bp0 + kb);   // 32B aligned (Kpad % 32 == 0)
    v16bf b1 = *(const v16bf*)(bp1 + kb);
    acc0 = __builtin_amdgcn_wmma_f32_16x16x32_bf16(
        false, a, false, b0, (short)0, acc0, false, false);
    acc1 = __builtin_amdgcn_wmma_f32_16x16x32_bf16(
        false, a, false, b1, (short)0, acc1, false, false);
  }
  // D: element r -> row i0 + r + half*8.
  float* op0 = Out + (size_t)(i0 + half * 8) * M + jcol;
#pragma unroll
  for (int r = 0; r < 8; ++r) op0[(size_t)r * M] = acc0[r];
  float* op1 = op0 + 16;
#pragma unroll
  for (int r = 0; r < 8; ++r) op1[(size_t)r * M] = acc1[r];
}

// ---------------- edge kernels ---------------------------------------------

// alpha[e,h] = sum_c att[h,c] * leaky_relu(xl[src,h,c] + xr[dst,h,c])
// One block (8 waves) per edge, one wave per head row, lanes stride over c.
__global__ void k_edge_alpha(const float* __restrict__ xl, const float* __restrict__ xr,
                             const int* __restrict__ src, const int* __restrict__ dst,
                             const float* __restrict__ att,
                             float* __restrict__ alpha, float* __restrict__ amax,
                             int cout) {
  const int e = blockIdx.x;
  const int s = src[e], d = dst[e];
  const int lane = threadIdx.x & 31;
  const int wave = threadIdx.x >> 5;
  for (int h = wave; h < GAT_H; h += 8) {
    const float* pl = xl + ((size_t)s * GAT_H + h) * cout;
    const float* pr = xr + ((size_t)d * GAT_H + h) * cout;
    const float* pa = att + (size_t)h * cout;
    float sum = 0.0f;
    for (int c = lane; c < cout; c += 32) {
      float v = pl[c] + pr[c];
      v = (v > 0.0f) ? v : NEG_SLOPE * v;
      sum = fmaf(pa[c], v, sum);
    }
#pragma unroll
    for (int off = 16; off > 0; off >>= 1) sum += __shfl_xor(sum, off, 32);
    if (lane == 0) {
      alpha[(size_t)e * GAT_H + h] = sum;
      atomicMax(&amax[(size_t)d * GAT_H + h], sum);
    }
  }
}

// ea[e,h] = exp(alpha - amax[dst,h]); denom[dst,h] += ea
__global__ void k_edge_exp(const float* __restrict__ alpha, const int* __restrict__ dst,
                           const float* __restrict__ amax,
                           float* __restrict__ ea, float* __restrict__ denom) {
  int idx = blockIdx.x * blockDim.x + threadIdx.x;
  if (idx >= GAT_E * GAT_H) return;
  int e = idx >> 5, h = idx & 31;
  int d = dst[e];
  float v = __expf(alpha[idx] - amax[(size_t)d * GAT_H + h]);
  ea[idx] = v;
  atomicAdd(&denom[(size_t)d * GAT_H + h], v);
}

// agg[dst,c] += (1/H) * sum_h (ea/denom) * xl[src,h,c]   (head-reduce first)
__global__ void k_edge_agg(const float* __restrict__ xl, const float* __restrict__ ea,
                           const float* __restrict__ denom,
                           const int* __restrict__ src, const int* __restrict__ dst,
                           float* __restrict__ agg, int cout) {
  __shared__ float w[GAT_H];
  const int e = blockIdx.x;
  const int s = src[e], d = dst[e];
  if (threadIdx.x < GAT_H) {
    float dn = denom[(size_t)d * GAT_H + threadIdx.x] + 1e-16f;
    w[threadIdx.x] = ea[(size_t)e * GAT_H + threadIdx.x] / dn * (1.0f / (float)GAT_H);
  }
  __syncthreads();
  const float* base = xl + (size_t)s * GAT_H * cout;
  for (int c = threadIdx.x; c < cout; c += blockDim.x) {
    float acc = 0.0f;
#pragma unroll 8
    for (int h = 0; h < GAT_H; ++h) acc = fmaf(w[h], base[(size_t)h * cout + c], acc);
    atomicAdd(&agg[(size_t)d * cout + c], acc);
  }
}

// epilogue: h = act(agg + b); also emit bf16 copy for next layer's GEMM.
__global__ void k_post(const float* __restrict__ agg, const float* __restrict__ b,
                       float* __restrict__ hout, __bf16* __restrict__ hbf,
                       float* __restrict__ dout, int cout, int act) {
  int idx = blockIdx.x * blockDim.x + threadIdx.x;
  if (idx >= GAT_N * cout) return;
  int c = idx % cout;
  float v = agg[idx] + b[c];
  v = (act == 0) ? fmaxf(v, 0.0f) : 1.0f / (1.0f + __expf(-v));
  if (dout) {
    dout[idx] = v;
  } else {
    hout[idx] = v;
    hbf[idx] = (__bf16)v;   // cout is a multiple of 32 -> next layer Kpad = cout
  }
}

// ---------------------------------------------------------------------------

extern "C" void kernel_launch(void* const* d_in, const int* in_sizes, int n_in,
                              void* d_out, int out_size, void* d_ws, size_t ws_size,
                              hipStream_t stream) {
  (void)in_sizes; (void)n_in; (void)out_size; (void)ws_size;

  static const int CIN[8]  = {4, 128, 256, 512, 1024, 512, 256, 128};
  static const int COUT[8] = {128, 256, 512, 1024, 512, 256, 128, 1};

  const float* x  = (const float*)d_in[0];
  const int*   ei = (const int*)d_in[1];
  const int* src = ei;            // edge_index[0]
  const int* dst = ei + GAT_E;    // edge_index[1]

  // workspace carve-up
  char* ws = (char*)d_ws;
  size_t off = 0;
  auto carve = [&](size_t bytes) { void* p = ws + off; off += (bytes + 255) & ~(size_t)255; return p; };
  float*  XL    = (float*)carve((size_t)GAT_N * 32768 * 4);  // 128 MB
  float*  XR    = (float*)carve((size_t)GAT_N * 32768 * 4);  // 128 MB
  float*  HF    = (float*)carve((size_t)GAT_N * 1024 * 4);
  __bf16* HB    = (__bf16*)carve((size_t)GAT_N * 1024 * 2);
  __bf16* WLT   = (__bf16*)carve((size_t)16777216 * 2);      // 32 MB (max Kpad*M)
  __bf16* WRT   = (__bf16*)carve((size_t)16777216 * 2);
  float*  ALPHA = (float*)carve((size_t)GAT_E * GAT_H * 4);
  float*  EAB   = (float*)carve((size_t)GAT_E * GAT_H * 4);
  float*  AMAX  = (float*)carve((size_t)GAT_N * GAT_H * 4);
  float*  DENOM = (float*)carve((size_t)GAT_N * GAT_H * 4);
  float*  AGG   = (float*)carve((size_t)GAT_N * 1024 * 4);

  // layer-1 input: pad K 4 -> 32, convert to bf16
  {
    int elems = GAT_N * 32;
    k_cvt_pad_cols<<<(elems + 255) / 256, 256, 0, stream>>>(x, HB, GAT_N, 4, 32);
  }

  for (int li = 0; li < 8; ++li) {
    const int cin  = CIN[li];
    const int cout = COUT[li];
    const int Kpad = (cin + 31) & ~31;
    const int M    = GAT_H * cout;
    const float* Wl  = (const float*)d_in[2 + 4 * li + 0];
    const float* Wr  = (const float*)d_in[2 + 4 * li + 1];
    const float* att = (const float*)d_in[2 + 4 * li + 2];
    const float* bia = (const float*)d_in[2 + 4 * li + 3];

    // weights -> bf16, transposed to WT[M x Kpad] (zero-pad K)
    {
      dim3 tg(M / 32, Kpad / 32);
      k_cvt_w_transpose<<<tg, 256, 0, stream>>>(Wl, WLT, cin, Kpad, M);
      k_cvt_w_transpose<<<tg, 256, 0, stream>>>(Wr, WRT, cin, Kpad, M);
    }

    // XL = H @ Wl, XR = H @ Wr via bf16 WMMA
    {
      dim3 grid((M + 127) / 128, GAT_N / 16);
      k_gemm_bf16<<<grid, 128, 0, stream>>>(HB, WLT, XL, Kpad, M);
      k_gemm_bf16<<<grid, 128, 0, stream>>>(HB, WRT, XR, Kpad, M);
    }

    // init reductions
    k_fill<<<(GAT_N * GAT_H + 255) / 256, 256, 0, stream>>>(AMAX, -3.0e38f, GAT_N * GAT_H);
    k_fill<<<(GAT_N * GAT_H + 255) / 256, 256, 0, stream>>>(DENOM, 0.0f, GAT_N * GAT_H);
    k_fill<<<(GAT_N * cout + 255) / 256, 256, 0, stream>>>(AGG, 0.0f, GAT_N * cout);

    // attention logits + segment max
    k_edge_alpha<<<GAT_E, 256, 0, stream>>>(XL, XR, src, dst, att, ALPHA, AMAX, cout);
    // exp + segment sum
    k_edge_exp<<<(GAT_E * GAT_H + 255) / 256, 256, 0, stream>>>(ALPHA, dst, AMAX, EAB, DENOM);
    // head-reduced weighted scatter
    k_edge_agg<<<GAT_E, 256, 0, stream>>>(XL, EAB, DENOM, src, dst, AGG, cout);

    // bias + activation (+ bf16 for next layer / final output)
    {
      int elems = GAT_N * cout;
      float* dout = (li == 7) ? (float*)d_out : nullptr;
      int act = (li == 7) ? 1 : 0;
      k_post<<<(elems + 255) / 256, 256, 0, stream>>>(AGG, bia, HF, HB, dout, cout, act);
    }
  }
}
